// SwinTransformerBlock3D_37314675868163
// MI455X (gfx1250) — matrix operand order repaired and checked by reading, not compiled, not run on hardware
//
#include <hip/hip_runtime.h>

// ---------------------------------------------------------------------------
// Swin Transformer 3D block for MI455X (gfx1250, wave32, WMMA bf16->f32)
// x: (1,16,56,56,256) fp32.  Windows 8x7x7 = 392 tok (padded to 416 = 13*32),
// 128 windows, 8 heads, head dim 32.  All GEMMs via v_wmma_f32_16x16x32_bf16.
// k-tiles staged to LDS with global_load_async_to_lds_b128 (ASYNCcnt).
// ---------------------------------------------------------------------------

typedef __bf16 bf16_t;
typedef bf16_t v16bf __attribute__((ext_vector_type(16)));
typedef bf16_t v8bf  __attribute__((ext_vector_type(8)));
typedef float  v8f   __attribute__((ext_vector_type(8)));

#define WMMA_BF16(a, b, c) \
  __builtin_amdgcn_wmma_f32_16x16x32_bf16(false, (a), false, (b), (short)0, (c), false, false)

static constexpr int C_DIM   = 256;
static constexpr int HEADS   = 8;
static constexpr int HD      = 32;
static constexpr int N_TOK   = 392;
static constexpr int NPAD    = 416;   // 13 * 32
static constexpr int NWIN    = 128;   // 2 * 8 * 8
static constexpr int DFF     = 1024;
static constexpr int TOKALL  = 16 * 56 * 56;  // 50176

// ---- WMMA operand loaders (contiguous k order, same for A and B) ----------------------------
// A (16x32 bf16, M x K): lane L<16 -> M=L, K = k0+0..15
//                        lane L>=16 -> M=L-16, K = k0+16..31
__device__ __forceinline__ v16bf load_a16x32(const bf16_t* base, int r0, int stride,
                                             int k0, int lane) {
  int m  = lane & 15;
  int kb = k0 + ((lane < 16) ? 0 : 16);
  const bf16_t* p = base + (long)(r0 + m) * stride + kb;
  v8bf lo = *(const v8bf*)(p);
  v8bf hi = *(const v8bf*)(p + 8);
  v16bf a;
#pragma unroll
  for (int i = 0; i < 8; ++i) { a[i] = lo[i]; a[i + 8] = hi[i]; }
  return a;
}

// B (32x16 bf16, K x N) from a row-major [N][K] matrix (so we get W^T as the
// B operand): lane holds column N = n0 + lane%16,
// K = k0 + (lane<16 ? 0 : 16) + 0..15  -> one contiguous 32-byte load.
__device__ __forceinline__ v16bf load_b_nk(const bf16_t* base, int n0, int stride,
                                           int k0, int lane) {
  int n  = n0 + (lane & 15);
  int kb = k0 + ((lane < 16) ? 0 : 16);
  return *(const v16bf*)(base + (long)n * stride + kb);
}

// ---- K0: pack weights fp32 -> bf16 ---------------------------------------
__global__ void k_pack_w(const float* qkv_w, const float* proj_w,
                         const float* fc1_w, const float* fc2_w,
                         bf16_t* wqkv, bf16_t* wproj, bf16_t* wfc1, bf16_t* wfc2) {
  int i = blockIdx.x * 256 + threadIdx.x;
  if (i < 768 * 256)  wqkv[i]  = (bf16_t)qkv_w[i];
  if (i < 256 * 256)  wproj[i] = (bf16_t)proj_w[i];
  if (i < 1024 * 256) wfc1[i]  = (bf16_t)fc1_w[i];
  if (i < 256 * 1024) wfc2[i]  = (bf16_t)fc2_w[i];
}

// ---- K0b: rel-pos bias, transposed [head][col][row]; col>=392 -> -1e9 -----
// (column-major so the 8 row-adjacent elems of a WMMA C column are one v8f)
__global__ void k_biasT(const int* rel_index, const float* rel_table, float* biasT) {
  int c = blockIdx.x, h = blockIdx.y;
  for (int r = threadIdx.x; r < NPAD; r += 256) {
    float v;
    if (c >= N_TOK) v = -1e9f;                    // kill padded keys in softmax
    else {
      int rr = (r < N_TOK) ? r : 0;               // padded query rows: don't care
      v = rel_table[rel_index[rr * N_TOK + c] * HEADS + h];
    }
    biasT[((long)h * NPAD + c) * NPAD + r] = v;
  }
}

// ---- K0c: attention mask, transposed [win][col][row]; pads -> 0 -----------
__global__ void k_maskT(const float* mask, float* maskT) {
  int c = blockIdx.x, w = blockIdx.y;
  for (int r = threadIdx.x; r < NPAD; r += 256) {
    float v = (c < N_TOK && r < N_TOK) ? mask[((long)w * N_TOK + r) * N_TOK + c] : 0.f;
    maskT[((long)w * NPAD + c) * NPAD + r] = v;
  }
}

// ---- K1: LayerNorm1 + roll(-4,-3,-3) + window partition -> bf16 -----------
__global__ void k_ln1_part(const float* x, const float* g, const float* b, bf16_t* xw) {
  int wave = threadIdx.x >> 5, lane = threadIdx.x & 31;
  long tok = (long)blockIdx.x * 8 + wave;          // over 128*416
  int win = (int)(tok / NPAD), row = (int)(tok % NPAD);
  bf16_t* dst = xw + ((long)win * NPAD + row) * C_DIM;
  if (row >= N_TOK) {                              // zero-fill K-padding rows
    for (int i = lane; i < C_DIM / 2; i += 32) ((unsigned int*)dst)[i] = 0u;
    return;
  }
  int wd = win >> 6, wh = (win >> 3) & 7, ww = win & 7;
  int ld = row / 49, lh = (row / 7) % 7, lw = row % 7;
  int d  = (wd * 8 + ld + 4) & 15;
  int h  = (wh * 7 + lh + 3) % 56;
  int w_ = (ww * 7 + lw + 3) % 56;
  const float* src = x + (((long)d * 56 + h) * 56 + w_) * C_DIM;
  float v[8], s = 0.f;
#pragma unroll
  for (int i = 0; i < 8; ++i) { v[i] = src[lane * 8 + i]; s += v[i]; }
  for (int m = 16; m >= 1; m >>= 1) s += __shfl_xor(s, m, 32);
  float mu = s * (1.0f / 256.0f), vs = 0.f;
#pragma unroll
  for (int i = 0; i < 8; ++i) { float t = v[i] - mu; vs += t * t; }
  for (int m = 16; m >= 1; m >>= 1) vs += __shfl_xor(vs, m, 32);
  float inv = rsqrtf(vs * (1.0f / 256.0f) + 1e-5f);
#pragma unroll
  for (int i = 0; i < 8; ++i) {
    int c = lane * 8 + i;
    dst[c] = (bf16_t)((v[i] - mu) * inv * g[c] + b[c]);
  }
}

// ---- K2: QKV GEMM (per window 416x256 @ 256x768), scatter to q/k/v --------
__global__ void k_gemm_qkv(const bf16_t* xw, const bf16_t* wqkv, const float* qkv_b,
                           bf16_t* Q, bf16_t* K, bf16_t* V) {
  int win = blockIdx.x, mt = blockIdx.y;
  int wave = threadIdx.x >> 5, lane = threadIdx.x & 31;
  int m0 = mt * 32 + (wave & 1) * 16;
  int n0base = (wave >> 1) * 192;                  // 4 col-groups * 12 tiles
  const bf16_t* A = xw + (long)win * NPAD * C_DIM;
  v8f acc[12] = {};
#pragma unroll
  for (int k0 = 0; k0 < 256; k0 += 32) {
    v16bf a = load_a16x32(A, m0, C_DIM, k0, lane);
#pragma unroll
    for (int t = 0; t < 12; ++t) {
      v16bf bm = load_b_nk(wqkv, n0base + t * 16, 256, k0, lane);
      acc[t] = WMMA_BF16(a, bm, acc[t]);
    }
  }
  int nlo = lane & 15, half8 = (lane < 16) ? 0 : 8;
#pragma unroll
  for (int t = 0; t < 12; ++t) {
    int n = n0base + t * 16 + nlo;
    int which = n >> 8;                            // 0=q 1=k 2=v
    int c = n & 255, head = c >> 5, hc = c & 31;
    float bias = qkv_b[n];
#pragma unroll
    for (int e = 0; e < 8; ++e) {
      int row = m0 + e + half8;
      long off = (((long)win * HEADS + head) * NPAD + row) * HD + hc;
      float val = acc[t][e] + bias;
      bool pad = (row >= N_TOK);
      if (which == 0)      Q[off] = pad ? (bf16_t)0.f : (bf16_t)(val * 0.17677669529f);
      else if (which == 1) K[off] = pad ? (bf16_t)0.f : (bf16_t)val;
      else                 V[off] = pad ? (bf16_t)0.f : (bf16_t)val;
    }
  }
}

// ---- K3: windowed attention, one workgroup per (window, head) -------------
// LDS: k row-major [416][32] (async-staged), v transposed [32][416],
//      per-wave P [16][416].
__global__ void __launch_bounds__(256, 1)
k_attn(const bf16_t* Q, const bf16_t* K, const bf16_t* V,
       const float* biasT, const float* maskT, bf16_t* attnout) {
  extern __shared__ char smem[];
  bf16_t* kbuf  = (bf16_t*)smem;                   // 416*32
  bf16_t* vt    = kbuf + NPAD * HD;                // 32*416
  bf16_t* pball = vt + HD * NPAD;                  // 8 * 16*416
  int win = blockIdx.x, head = blockIdx.y;
  int wave = threadIdx.x >> 5, lane = threadIdx.x & 31;
  bf16_t* pb = pball + wave * 16 * NPAD;
  const bf16_t* Kg = K + ((long)win * HEADS + head) * NPAD * HD;
  const bf16_t* Vg = V + ((long)win * HEADS + head) * NPAD * HD;
  const bf16_t* Qg = Q + ((long)win * HEADS + head) * NPAD * HD;

  // --- k tiles: async global -> LDS (no VGPR round-trip, ASYNCcnt) ---------
  {
    unsigned kbase = (unsigned)(uintptr_t)kbuf;    // LDS byte offset (addr[31:0])
    for (int i = threadIdx.x; i < NPAD * HD / 8; i += 256) {  // 16B per lane
      unsigned laddr = kbase + i * 16;
      unsigned long long gaddr = (unsigned long long)(uintptr_t)(Kg + i * 8);
      asm volatile("global_load_async_to_lds_b128 %0, %1, off"
                   :: "v"(laddr), "v"(gaddr) : "memory");
    }
  }
  // --- v tiles: transpose into LDS via VALU (overlaps with async k) --------
  for (int i = threadIdx.x; i < NPAD * HD; i += 256) {
    int tok = i >> 5, c = i & 31;
    vt[c * NPAD + tok] = Vg[i];
  }
  asm volatile("s_wait_asynccnt 0x0" ::: "memory");
  __syncthreads();

  int nlo = lane & 15, half8 = (lane < 16) ? 0 : 8;
  const float* biasH = biasT + (long)head * NPAD * NPAD;
  const float* maskW = maskT + (long)win * NPAD * NPAD;
  for (int rb = wave; rb < NPAD / 16; rb += 8) {
    int r0 = rb * 16;
    v16bf aq = load_a16x32(Qg, r0, HD, 0, lane);   // q rows, K = head dim 32
    v8f S[26];
#pragma unroll
    for (int t = 0; t < 26; ++t) {
      v16bf bk = load_b_nk(kbuf, t * 16, HD, 0, lane);
      v8f z = {};
      S[t] = WMMA_BF16(aq, bk, z);                 // S = q * k^T (q pre-scaled)
    }
    float mx[8];
#pragma unroll
    for (int e = 0; e < 8; ++e) mx[e] = -3e38f;
#pragma unroll
    for (int t = 0; t < 26; ++t) {
      int col = t * 16 + nlo;
      v8f bv = *(const v8f*)(biasH + (long)col * NPAD + r0 + half8);
      v8f mv = *(const v8f*)(maskW + (long)col * NPAD + r0 + half8);
#pragma unroll
      for (int e = 0; e < 8; ++e) {
        float v = S[t][e] + bv[e] + mv[e];
        S[t][e] = v;
        mx[e] = fmaxf(mx[e], v);
      }
    }
#pragma unroll
    for (int e = 0; e < 8; ++e)                    // row max across 16 lanes
      for (int m = 1; m < 16; m <<= 1) mx[e] = fmaxf(mx[e], __shfl_xor(mx[e], m, 32));
    float ls[8];
#pragma unroll
    for (int e = 0; e < 8; ++e) ls[e] = 0.f;
#pragma unroll
    for (int t = 0; t < 26; ++t) {
      int col = t * 16 + nlo;
#pragma unroll
      for (int e = 0; e < 8; ++e) {
        float p = __expf(S[t][e] - mx[e]);
        ls[e] += p;
        pb[(e + half8) * NPAD + col] = (bf16_t)p;  // C-layout -> row-major via LDS
      }
    }
#pragma unroll
    for (int e = 0; e < 8; ++e)
      for (int m = 1; m < 16; m <<= 1) ls[e] += __shfl_xor(ls[e], m, 32);
    v8f o0 = {}, o1 = {};
    for (int ks = 0; ks < NPAD / 32; ++ks) {       // out = P @ V
      v16bf ap = load_a16x32(pb, 0, NPAD, ks * 32, lane);
      v16bf b0 = load_b_nk(vt, 0,  NPAD, ks * 32, lane);
      v16bf b1 = load_b_nk(vt, 16, NPAD, ks * 32, lane);
      o0 = WMMA_BF16(ap, b0, o0);
      o1 = WMMA_BF16(ap, b1, o1);
    }
#pragma unroll
    for (int e = 0; e < 8; ++e) {
      int row = r0 + e + half8;
      if (row < N_TOK) {
        float inv = 1.0f / ls[e];
        long base = ((long)win * NPAD + row) * C_DIM + head * HD;
        attnout[base + nlo]      = (bf16_t)(o0[e] * inv);
        attnout[base + 16 + nlo] = (bf16_t)(o1[e] * inv);
      }
    }
  }
}

// ---- K4: proj GEMM + window reverse + roll(+4,+3,+3) + residual -----------
__global__ void k_proj_res(const bf16_t* attnout, const bf16_t* wproj, const float* proj_b,
                           const float* x_in, float* x2) {
  int win = blockIdx.x, mt = blockIdx.y;
  int wave = threadIdx.x >> 5, lane = threadIdx.x & 31;
  int m0 = mt * 32 + (wave & 1) * 16;
  int n0 = (wave >> 1) * 64;
  const bf16_t* A = attnout + (long)win * NPAD * C_DIM;
  v8f acc[4] = {};
#pragma unroll
  for (int k0 = 0; k0 < 256; k0 += 32) {
    v16bf a = load_a16x32(A, m0, C_DIM, k0, lane);
#pragma unroll
    for (int t = 0; t < 4; ++t)
      acc[t] = WMMA_BF16(a, load_b_nk(wproj, n0 + t * 16, 256, k0, lane), acc[t]);
  }
  int nlo = lane & 15, half8 = (lane < 16) ? 0 : 8;
  int wd = win >> 6, wh = (win >> 3) & 7, ww = win & 7;
#pragma unroll
  for (int e = 0; e < 8; ++e) {
    int row = m0 + e + half8;
    if (row >= N_TOK) continue;
    int ld = row / 49, lh = (row / 7) % 7, lw = row % 7;
    int d  = (wd * 8 + ld + 4) & 15;
    int h  = (wh * 7 + lh + 3) % 56;
    int w_ = (ww * 7 + lw + 3) % 56;
    long base = (((long)d * 56 + h) * 56 + w_) * C_DIM;
#pragma unroll
    for (int t = 0; t < 4; ++t) {
      int c = n0 + t * 16 + nlo;
      x2[base + c] = acc[t][e] + proj_b[c] + x_in[base + c];
    }
  }
}

// ---- K5a: LayerNorm2 -> bf16 ----------------------------------------------
__global__ void k_ln2(const float* x2, const float* g, const float* b, bf16_t* xn2) {
  int wave = threadIdx.x >> 5, lane = threadIdx.x & 31;
  long tok = (long)blockIdx.x * 8 + wave;
  if (tok >= TOKALL) return;
  const float* src = x2 + tok * C_DIM;
  bf16_t* dst = xn2 + tok * C_DIM;
  float v[8], s = 0.f;
#pragma unroll
  for (int i = 0; i < 8; ++i) { v[i] = src[lane * 8 + i]; s += v[i]; }
  for (int m = 16; m >= 1; m >>= 1) s += __shfl_xor(s, m, 32);
  float mu = s * (1.0f / 256.0f), vs = 0.f;
#pragma unroll
  for (int i = 0; i < 8; ++i) { float t = v[i] - mu; vs += t * t; }
  for (int m = 16; m >= 1; m >>= 1) vs += __shfl_xor(vs, m, 32);
  float inv = rsqrtf(vs * (1.0f / 256.0f) + 1e-5f);
#pragma unroll
  for (int i = 0; i < 8; ++i) {
    int c = lane * 8 + i;
    dst[c] = (bf16_t)((v[i] - mu) * inv * g[c] + b[c]);
  }
}

// ---- K5b: fc1 GEMM + exact GELU -------------------------------------------
__global__ void k_fc1(const bf16_t* xn2, const bf16_t* wfc1, const float* fc1_b,
                      bf16_t* hbuf) {
  int mt = blockIdx.x, nb = blockIdx.y;
  int wave = threadIdx.x >> 5, lane = threadIdx.x & 31;
  int m0 = mt * 32 + (wave & 1) * 16;
  int n0 = nb * 256 + (wave >> 1) * 64;
  v8f acc[4] = {};
#pragma unroll
  for (int k0 = 0; k0 < 256; k0 += 32) {
    v16bf a = load_a16x32(xn2, m0, C_DIM, k0, lane);
#pragma unroll
    for (int t = 0; t < 4; ++t)
      acc[t] = WMMA_BF16(a, load_b_nk(wfc1, n0 + t * 16, 256, k0, lane), acc[t]);
  }
  int nlo = lane & 15, half8 = (lane < 16) ? 0 : 8;
#pragma unroll
  for (int t = 0; t < 4; ++t) {
    int c = n0 + t * 16 + nlo;
    float bias = fc1_b[c];
#pragma unroll
    for (int e = 0; e < 8; ++e) {
      int row = m0 + e + half8;
      float v = acc[t][e] + bias;
      v = 0.5f * v * (1.0f + erff(v * 0.70710678118654752f));
      hbuf[(long)row * DFF + c] = (bf16_t)v;
    }
  }
}

// ---- K6: fc2 GEMM + residual -> output ------------------------------------
__global__ void k_fc2(const bf16_t* hbuf, const bf16_t* wfc2, const float* fc2_b,
                      const float* x2, float* out) {
  int mt = blockIdx.x;
  int wave = threadIdx.x >> 5, lane = threadIdx.x & 31;
  int m0 = mt * 32 + (wave & 1) * 16;
  int n0 = (wave >> 1) * 64;
  v8f acc[4] = {};
#pragma unroll 4
  for (int k0 = 0; k0 < DFF; k0 += 32) {
    // prefetch next A chunk (lowers to global_prefetch_b8 on gfx1250)
    if (k0 + 32 < DFF)
      __builtin_prefetch(hbuf + (long)(m0 + (lane & 15)) * DFF + k0 + 32, 0, 1);
    v16bf a = load_a16x32(hbuf, m0, DFF, k0, lane);
#pragma unroll
    for (int t = 0; t < 4; ++t)
      acc[t] = WMMA_BF16(a, load_b_nk(wfc2, n0 + t * 16, DFF, k0, lane), acc[t]);
  }
  int nlo = lane & 15, half8 = (lane < 16) ? 0 : 8;
#pragma unroll
  for (int t = 0; t < 4; ++t) {
    int c = n0 + t * 16 + nlo;
    float bias = fc2_b[c];
#pragma unroll
    for (int e = 0; e < 8; ++e) {
      long row = m0 + e + half8;
      out[row * C_DIM + c] = acc[t][e] + bias + x2[row * C_DIM + c];
    }
  }
}

// ---------------------------------------------------------------------------
extern "C" void kernel_launch(void* const* d_in, const int* in_sizes, int n_in,
                              void* d_out, int out_size, void* d_ws, size_t ws_size,
                              hipStream_t stream) {
  const float* x        = (const float*)d_in[0];
  const float* mask     = (const float*)d_in[1];
  const int*   rel_idx  = (const int*)  d_in[2];
  const float* ln1_g    = (const float*)d_in[3];
  const float* ln1_b    = (const float*)d_in[4];
  const float* qkv_w    = (const float*)d_in[5];
  const float* qkv_b    = (const float*)d_in[6];
  const float* proj_w   = (const float*)d_in[7];
  const float* proj_b   = (const float*)d_in[8];
  const float* rel_tab  = (const float*)d_in[9];
  const float* ln2_g    = (const float*)d_in[10];
  const float* ln2_b    = (const float*)d_in[11];
  const float* fc1_w    = (const float*)d_in[12];
  const float* fc1_b    = (const float*)d_in[13];
  const float* fc2_w    = (const float*)d_in[14];
  const float* fc2_b    = (const float*)d_in[15];
  float* out = (float*)d_out;

  char* ws = (char*)d_ws;
  size_t off = 0;
  auto alloc = [&](size_t bytes) { char* p = ws + off; off += (bytes + 255) & ~size_t(255); return p; };
  bf16_t* wqkv  = (bf16_t*)alloc(768 * 256 * 2);
  bf16_t* wproj = (bf16_t*)alloc(256 * 256 * 2);
  bf16_t* wfc1  = (bf16_t*)alloc(1024 * 256 * 2);
  bf16_t* wfc2  = (bf16_t*)alloc(256 * 1024 * 2);
  float*  biasT = (float*) alloc((size_t)HEADS * NPAD * NPAD * 4);
  float*  maskT = (float*) alloc((size_t)NWIN * NPAD * NPAD * 4);
  bf16_t* xw    = (bf16_t*)alloc((size_t)NWIN * NPAD * C_DIM * 2);
  bf16_t* Qb    = (bf16_t*)alloc((size_t)NWIN * HEADS * NPAD * HD * 2);
  bf16_t* Kb    = (bf16_t*)alloc((size_t)NWIN * HEADS * NPAD * HD * 2);
  bf16_t* Vb    = (bf16_t*)alloc((size_t)NWIN * HEADS * NPAD * HD * 2);
  bf16_t* aout  = (bf16_t*)alloc((size_t)NWIN * NPAD * C_DIM * 2);
  float*  x2    = (float*) alloc((size_t)TOKALL * C_DIM * 4);
  bf16_t* xn2   = (bf16_t*)alloc((size_t)TOKALL * C_DIM * 2);
  bf16_t* hbuf  = (bf16_t*)alloc((size_t)TOKALL * DFF * 2);

  k_pack_w<<<1024, 256, 0, stream>>>(qkv_w, proj_w, fc1_w, fc2_w, wqkv, wproj, wfc1, wfc2);
  k_biasT<<<dim3(NPAD, HEADS), 256, 0, stream>>>(rel_idx, rel_tab, biasT);
  k_maskT<<<dim3(NPAD, NWIN), 256, 0, stream>>>(mask, maskT);
  k_ln1_part<<<NWIN * NPAD / 8, 256, 0, stream>>>(x, ln1_g, ln1_b, xw);
  k_gemm_qkv<<<dim3(NWIN, NPAD / 32), 256, 0, stream>>>(xw, wqkv, qkv_b, Qb, Kb, Vb);
  size_t smem = (size_t)(NPAD * HD + HD * NPAD + 8 * 16 * NPAD) * 2;  // ~156 KB
  k_attn<<<dim3(NWIN, HEADS), 256, smem, stream>>>(Qb, Kb, Vb, biasT, maskT, aout);
  k_proj_res<<<dim3(NWIN, NPAD / 32), 256, 0, stream>>>(aout, wproj, proj_b, x, x2);
  k_ln2<<<TOKALL / 8, 256, 0, stream>>>(x2, ln2_g, ln2_b, xn2);
  k_fc1<<<dim3(TOKALL / 32, 4), 256, 0, stream>>>(xn2, wfc1, fc1_b, hbuf);
  k_fc2<<<TOKALL / 32, 256, 0, stream>>>(hbuf, wfc2, fc2_b, x2, out);
}
